// BioActivityGNN_5806795784761
// MI455X (gfx1250) — matrix operand order repaired and checked by reading, not compiled, not run on hardware
//
#include <hip/hip_runtime.h>
#include <hip/hip_bf16.h>

typedef __attribute__((ext_vector_type(16))) _Float16 v16h;
typedef __attribute__((ext_vector_type(8)))  float    v8f;

#define TPB 256
#define NT_BLK 4   // N-tiles per wave (16x64 output strip)

// ---------------------------------------------------------------------------
// Utility kernels
// ---------------------------------------------------------------------------

__global__ void k_fill_f32(float* p, float v, long long n) {
    long long i = blockIdx.x * (long long)blockDim.x + threadIdx.x;
    if (i < n) p[i] = v;
}

__global__ void k_f32_to_f16(const float* __restrict__ in, _Float16* __restrict__ out, long long n) {
    long long i = blockIdx.x * (long long)blockDim.x + threadIdx.x;
    if (i < n) out[i] = (_Float16)in[i];
}

// deg[dst] += 1 per edge (deg pre-initialized to 1.0 for self-loops)
__global__ void k_deg_accum(const int* __restrict__ dst, float* __restrict__ deg, int E) {
    int e = blockIdx.x * blockDim.x + threadIdx.x;
    if (e < E) atomicAdd(&deg[dst[e]], 1.0f);
}

__global__ void k_dinv(float* __restrict__ deg, int N) {
    int i = blockIdx.x * blockDim.x + threadIdx.x;
    if (i < N) deg[i] = rsqrtf(deg[i]);   // deg >= 1 always (self-loops)
}

__global__ void k_norm(const int* __restrict__ src, const int* __restrict__ dst,
                       const float* __restrict__ dinv, float* __restrict__ norm, int E) {
    int e = blockIdx.x * blockDim.x + threadIdx.x;
    if (e < E) norm[e] = dinv[src[e]] * dinv[dst[e]];
}

// ---------------------------------------------------------------------------
// Weight packing: row-major f32 W[K][Nout] -> per-lane WMMA B-fragment layout.
// B (32x16 f16): lane l in [0,16): col=l, holds K=0..7 as e=K, K=16..23 as e=K-8;
//                lane l+16:        col=l, holds K=8..15 as e=K-8, K=24..31 as e=K-16.
// Packed address: ((ktile*ntiles + ntile)*32 + lane)*16 + e   (contiguous 32B/lane)
// ---------------------------------------------------------------------------
__global__ void k_pack_w(const float* __restrict__ W, _Float16* __restrict__ Bp,
                         int K, int Nout) {
    int idx = blockIdx.x * blockDim.x + threadIdx.x;
    if (idx >= K * Nout) return;
    int k = idx / Nout, n = idx % Nout;
    int kt = k >> 5, kk = k & 31;
    int nt = n >> 4, col = n & 15;
    int ntiles = Nout >> 4;
    int lane = col + ((kk & 8) ? 16 : 0);
    int e    = (kk & 7) + ((kk & 16) ? 8 : 0);
    Bp[(((size_t)kt * ntiles + nt) * 32 + lane) * 16 + e] = (_Float16)W[(size_t)k * Nout + n];
}

// ---------------------------------------------------------------------------
// WMMA GEMM:  C[N x Nout] = A[N x K] (f16 row-major) * Bpacked  (f32 out)
// Each wave computes a 16 x (16*NT_BLK) strip of C.  Fully unrolled over the
// KC K-chunks (KC = K/32 is a template parameter: 2/4/8).  Per chunk: one A
// fragment load + NT_BLK B fragments preloaded into distinct registers, then
// an NT_BLK WMMA burst (loads of the next fragments overlap the burst).
// A fragment (16-bit 16x32, ISA layout): lane l: row = l&15;
//   lanes 0-15 hold K pairs {0,1},{2,3},{4,5},{6,7},{16,17},...,{22,23}
//   lanes 16-31: same +8.
// ---------------------------------------------------------------------------
template <int KC>
__global__ void k_gemm_wmma(const _Float16* __restrict__ A,
                            const _Float16* __restrict__ Bp,
                            float* __restrict__ C,
                            int N, int Nout) {
    const int K = KC * 32;
    int wave = (blockIdx.x * blockDim.x + threadIdx.x) >> 5;
    int lane = threadIdx.x & 31;
    int mtiles = N >> 4, ntiles = Nout >> 4;
    int ngroups = ntiles / NT_BLK;
    if (wave >= mtiles * ngroups) return;
    int mt  = wave / ngroups;
    int nt0 = (wave % ngroups) * NT_BLK;
    int row = lane & 15;
    int hi  = lane >> 4;            // 0 for lanes 0-15, 1 for lanes 16-31
    int kofs = hi ? 8 : 0;

    v8f acc[NT_BLK];
#pragma unroll
    for (int j = 0; j < NT_BLK; ++j) acc[j] = (v8f){};

    const _Float16* arow  = A + (size_t)(mt * 16 + row) * K + kofs;
    const _Float16* bbase = Bp + ((size_t)nt0 * 32 + lane) * 16;

#pragma unroll
    for (int t = 0; t < KC; ++t) {
        union { v16h v; unsigned u[8]; } a;
        const _Float16* ap = arow + t * 32;
#pragma unroll
        for (int v = 0; v < 8; ++v) {
            int kk = (v < 4) ? (2 * v) : (16 + 2 * (v - 4));
            a.u[v] = *(const unsigned*)(ap + kk);   // even offset -> 4B aligned
        }
        // preload all NT_BLK B fragments into distinct registers so the
        // WMMA burst below can overlap outstanding loads
        const uint4* bk = (const uint4*)(bbase + (size_t)t * ntiles * 32 * 16);
        union { v16h v; uint4 q[2]; } b[NT_BLK];
#pragma unroll
        for (int j = 0; j < NT_BLK; ++j) {
            b[j].q[0] = bk[(size_t)j * 32 * 2 + 0];   // next ntile = 512 halves away
            b[j].q[1] = bk[(size_t)j * 32 * 2 + 1];
        }
#pragma unroll
        for (int j = 0; j < NT_BLK; ++j) {
            acc[j] = __builtin_amdgcn_wmma_f32_16x16x32_f16(
                /*neg_a=*/false, a.v, /*neg_b=*/false, b[j].v,
                /*c_mod=*/(short)0, acc[j], /*reuse_a=*/false, /*reuse_b=*/false);
        }
    }

    // C/D layout: lanes 0-15: N=lane, M=r; lanes 16-31: N=lane-16, M=8+r
    int outRow = mt * 16 + (hi ? 8 : 0);
#pragma unroll
    for (int j = 0; j < NT_BLK; ++j) {
        int col = (nt0 + j) * 16 + row;
#pragma unroll
        for (int r = 0; r < 8; ++r)
            C[(size_t)(outRow + r) * Nout + col] = acc[j][r];
    }
}

// ---------------------------------------------------------------------------
// Per-node seed: agg[i][c] = t[i][c] * dinv[i]^2 (self-loop) + b[c]
// ---------------------------------------------------------------------------
__global__ void k_seed_agg(const float* __restrict__ t, const float* __restrict__ dinv,
                           const float* __restrict__ bias, float* __restrict__ agg,
                           int N, int shiftF) {
    long long idx = blockIdx.x * (long long)blockDim.x + threadIdx.x;
    int F = 1 << shiftF;
    if (idx >= ((long long)N << shiftF)) return;
    int i = (int)(idx >> shiftF);
    int c = (int)(idx & (F - 1));
    float d = dinv[i];
    agg[idx] = t[idx] * d * d + bias[c];
}

// ---------------------------------------------------------------------------
// Edge scatter: agg[dst] += t[src] * norm   (float4 per thread, 4 atomics)
// ---------------------------------------------------------------------------
__global__ void k_edge_scatter(const int* __restrict__ src, const int* __restrict__ dst,
                               const float* __restrict__ norm, const float* __restrict__ t,
                               float* __restrict__ agg, int E, int shiftQ) {
    long long idx = blockIdx.x * (long long)blockDim.x + threadIdx.x;
    if (idx >= ((long long)E << shiftQ)) return;
    int e  = (int)(idx >> shiftQ);
    int c4 = (int)(idx & ((1 << shiftQ) - 1));
    int F  = 4 << shiftQ;
    float w = norm[e];
    const float4 v = *(const float4*)(t + (size_t)src[e] * F + c4 * 4);
    float* o = agg + (size_t)dst[e] * F + c4 * 4;
    atomicAdd(o + 0, v.x * w);
    atomicAdd(o + 1, v.y * w);
    atomicAdd(o + 2, v.z * w);
    atomicAdd(o + 3, v.w * w);
}

// relu + convert to f16 (GEMM input for next layer)
__global__ void k_relu_to_f16(const float* __restrict__ agg, _Float16* __restrict__ acth,
                              long long n) {
    long long i = blockIdx.x * (long long)blockDim.x + threadIdx.x;
    if (i < n) acth[i] = (_Float16)fmaxf(agg[i], 0.0f);
}

// ---------------------------------------------------------------------------
// Pooling
// ---------------------------------------------------------------------------
__global__ void k_pool_cnt(const int* __restrict__ batch, float* __restrict__ cnts, int N) {
    int i = blockIdx.x * blockDim.x + threadIdx.x;
    if (i < N) atomicAdd(&cnts[batch[i]], 1.0f);
}

__global__ void k_pool_accum(const int* __restrict__ batch, const float* __restrict__ h,
                             float* __restrict__ sums, int N) {
    long long idx = blockIdx.x * (long long)blockDim.x + threadIdx.x;   // N * 64 quarters
    if (idx >= (long long)N * 64) return;
    int i  = (int)(idx >> 6);
    int c4 = (int)(idx & 63);
    int g = batch[i];
    const float4 v = *(const float4*)(h + (size_t)i * 256 + c4 * 4);
    float* o = sums + (size_t)g * 256 + c4 * 4;
    atomicAdd(o + 0, v.x);
    atomicAdd(o + 1, v.y);
    atomicAdd(o + 2, v.z);
    atomicAdd(o + 3, v.w);
}

__global__ void k_final(const float* __restrict__ sums, const float* __restrict__ cnts,
                        const float* __restrict__ linW, const float* __restrict__ linb,
                        float* __restrict__ out, int G) {
    int g = blockIdx.x * blockDim.x + threadIdx.x;
    if (g >= G) return;
    float inv = 1.0f / fmaxf(cnts[g], 1.0f);
    float acc = 0.0f;
    for (int k = 0; k < 256; ++k) acc += sums[(size_t)g * 256 + k] * inv * linW[k];
    out[g] = acc + linb[0];
}

// ---------------------------------------------------------------------------
// Host launch
// ---------------------------------------------------------------------------
static inline long long ceil_div(long long a, long long b) { return (a + b - 1) / b; }

extern "C" void kernel_launch(void* const* d_in, const int* in_sizes, int n_in,
                              void* d_out, int out_size, void* d_ws, size_t ws_size,
                              hipStream_t stream) {
    const float* x    = (const float*)d_in[0];      // [N,64]
    const int*   ei   = (const int*)d_in[1];        // [2,E]
    const int*   bat  = (const int*)d_in[2];        // [N]
    const float* W1   = (const float*)d_in[3];      // [64,128]
    const float* b1   = (const float*)d_in[4];      // [128]
    const float* W2   = (const float*)d_in[5];      // [128,256]
    const float* b2   = (const float*)d_in[6];      // [256]
    const float* W3   = (const float*)d_in[7];      // [256,256]
    const float* b3   = (const float*)d_in[8];      // [256]
    const float* linW = (const float*)d_in[9];      // [256,1]
    const float* linb = (const float*)d_in[10];     // [1]
    float* out = (float*)d_out;

    const int N = in_sizes[2];          // 50000 (multiple of 16)
    const int E = in_sizes[1] / 2;      // 1000000
    const int FIN = 64, H = 128, H2 = 256;
    const int G = out_size;             // 128 graphs

    const int* src = ei;
    const int* dst = ei + E;

    // ---- workspace carve-out (256B aligned) ----
    char* p = (char*)d_ws;
    auto carve = [&](size_t bytes) {
        char* r = p;
        p += (bytes + 255) & ~(size_t)255;
        return r;
    };
    float*    t_buf  = (float*)   carve((size_t)N * H2 * 4);   // GEMM output / messages
    float*    agg    = (float*)   carve((size_t)N * H2 * 4);   // aggregation target
    _Float16* acth   = (_Float16*)carve((size_t)N * H2 * 2);   // f16 activations (GEMM A)
    float*    deg    = (float*)   carve((size_t)N * 4);        // degree -> dinv (in place)
    float*    norm   = (float*)   carve((size_t)E * 4);
    _Float16* Wp1    = (_Float16*)carve((size_t)FIN * H  * 2);
    _Float16* Wp2    = (_Float16*)carve((size_t)H   * H2 * 2);
    _Float16* Wp3    = (_Float16*)carve((size_t)H2  * H2 * 2);
    float*    sums   = (float*)   carve((size_t)G * H2 * 4);
    float*    cnts   = (float*)   carve((size_t)G * 4);

    // ---- pack weights to WMMA B-fragment layout ----
    k_pack_w<<<ceil_div(FIN * H, TPB), TPB, 0, stream>>>(W1, Wp1, FIN, H);
    k_pack_w<<<ceil_div(H * H2, TPB), TPB, 0, stream>>>(W2, Wp2, H, H2);
    k_pack_w<<<ceil_div(H2 * H2, TPB), TPB, 0, stream>>>(W3, Wp3, H2, H2);

    // ---- gcn_norm: deg (with self-loops), dinv, edge norms ----
    k_fill_f32<<<ceil_div(N, TPB), TPB, 0, stream>>>(deg, 1.0f, N);
    k_deg_accum<<<ceil_div(E, TPB), TPB, 0, stream>>>(dst, deg, E);
    k_dinv<<<ceil_div(N, TPB), TPB, 0, stream>>>(deg, N);          // deg now holds dinv
    k_norm<<<ceil_div(E, TPB), TPB, 0, stream>>>(src, dst, deg, norm, E);

    // ---- x -> f16 ----
    k_f32_to_f16<<<ceil_div((long long)N * FIN, TPB), TPB, 0, stream>>>(x, acth, (long long)N * FIN);

    const long long gw1 = (long long)(N >> 4) * ((H  >> 4) / NT_BLK) * 32;
    const long long gw2 = (long long)(N >> 4) * ((H2 >> 4) / NT_BLK) * 32;

    // ================= Layer 1: 64 -> 128, relu =================
    {
        k_gemm_wmma<2><<<ceil_div(gw1, TPB), TPB, 0, stream>>>(acth, Wp1, t_buf, N, H);
        k_seed_agg<<<ceil_div((long long)N * H, TPB), TPB, 0, stream>>>(t_buf, deg, b1, agg, N, 7);
        k_edge_scatter<<<ceil_div((long long)E * (H / 4), TPB), TPB, 0, stream>>>(src, dst, norm, t_buf, agg, E, 5);
        k_relu_to_f16<<<ceil_div((long long)N * H, TPB), TPB, 0, stream>>>(agg, acth, (long long)N * H);
    }
    // ================= Layer 2: 128 -> 256, relu =================
    {
        k_gemm_wmma<4><<<ceil_div(gw2, TPB), TPB, 0, stream>>>(acth, Wp2, t_buf, N, H2);
        k_seed_agg<<<ceil_div((long long)N * H2, TPB), TPB, 0, stream>>>(t_buf, deg, b2, agg, N, 8);
        k_edge_scatter<<<ceil_div((long long)E * (H2 / 4), TPB), TPB, 0, stream>>>(src, dst, norm, t_buf, agg, E, 6);
        k_relu_to_f16<<<ceil_div((long long)N * H2, TPB), TPB, 0, stream>>>(agg, acth, (long long)N * H2);
    }
    // ================= Layer 3: 256 -> 256, no relu =================
    {
        k_gemm_wmma<8><<<ceil_div(gw2, TPB), TPB, 0, stream>>>(acth, Wp3, t_buf, N, H2);
        k_seed_agg<<<ceil_div((long long)N * H2, TPB), TPB, 0, stream>>>(t_buf, deg, b3, agg, N, 8);
        k_edge_scatter<<<ceil_div((long long)E * (H2 / 4), TPB), TPB, 0, stream>>>(src, dst, norm, t_buf, agg, E, 6);
        // agg now holds h3 (f32), bias included
    }

    // ================= Global mean pool + linear head =================
    k_fill_f32<<<ceil_div((long long)G * H2 + G, TPB), TPB, 0, stream>>>(sums, 0.0f, (long long)G * H2 + G);
    k_fill_f32<<<1, TPB, 0, stream>>>(cnts, 0.0f, G);
    k_pool_cnt<<<ceil_div(N, TPB), TPB, 0, stream>>>(bat, cnts, N);
    k_pool_accum<<<ceil_div((long long)N * 64, TPB), TPB, 0, stream>>>(bat, agg, sums, N);
    k_final<<<ceil_div(G, TPB), TPB, 0, stream>>>(sums, cnts, linW, linb, out, G);
}
